// Gemma_90443421319540
// MI455X (gfx1250) — matrix operand order repaired and checked by reading, not compiled, not run on hardware
//
#include <hip/hip_runtime.h>
#include <math.h>

// ---------------- model constants ----------------
#define Dm   2048
#define Hn   8
#define HDs  256
#define KVs  4
#define GHD  512
#define GKV  4
#define ROTg 128
#define Fm   8192
#define Vm   32000
#define Bm_  2
#define Sm   1024
#define WINs 512
#define EPSf 1e-6f

#define MROWS (Bm_ * Sm)   // 2048 token rows

// ---------------- WMMA types ----------------
typedef __attribute__((ext_vector_type(16))) __bf16 bf16x16;
typedef __attribute__((ext_vector_type(8)))  __bf16 bf16x8;
typedef __attribute__((ext_vector_type(8)))  float  floatx8;

union bfrag_u { bf16x16 v; bf16x8 h[2]; };

// =====================================================================
// Tiled bf16 WMMA GEMM:  C[M,N] = A[M,K] * B[K,N]   (fp32 in, fp32 out)
// Block tile 128x128 (8 waves as 4x2); wave tile 32x64 = 2x4 WMMA tiles,
// 8 x v_wmma_f32_16x16x32_bf16 per K-step (K-step = 32).
//
// LDS is laid out in *fragment order* so every per-lane fragment gather
// is 32 contiguous bytes -> 2x ds_load_b128:
//   As[m][.] rows hold K in contiguous order [K0-7, K8-15, K16-23, K24-31];
//     lane<16 needs K0-15 = positions 0..15,
//     lane>=16 needs K16-31 = positions 16..31.
//   Bs[n][k] is the B tile transposed; fragment = Bs[n][(lane/16)*16 + i].
// Row stride 40 bf16 (80 B) keeps 16-B alignment and staggers banks.
//
// btrans==1: logical B(k,n) = Bsrc[n*ldb + k]  (tied-embedding logits).
// Assumes M%128==0, N%128==0, K%32==0 (true for every GEMM here).
// =====================================================================
__global__ __launch_bounds__(256) void gemm_bf16_wmma_kernel(
    const float* __restrict__ A, const float* __restrict__ Bsrc,
    float* __restrict__ C, int M, int N, int K,
    int lda, int ldb, int ldc, int btrans)
{
    __shared__ __bf16 As[128][40];   // 10 KB
    __shared__ __bf16 Bs[128][40];   // 10 KB (transposed: [n][k])

    const int tid   = threadIdx.x;
    const int wave  = tid >> 5;
    const int lane  = tid & 31;
    const int wm    = (wave >> 1) << 5;   // 0,32,64,96
    const int wn    = (wave & 1) << 6;    // 0,64
    const int m0    = blockIdx.y * 128;
    const int n0    = blockIdx.x * 128;
    const int lrow  = lane & 15;
    const int lhalf = (lane >> 4) << 4;   // element offset 0 or 16

    floatx8 acc[2][4] = {};

    for (int kb = 0; kb < K; kb += 32) {
        // ---- stage A (128x32): thread t -> row m = t>>1, sel = t&1 (k 0..15 / 16..31)
        {
            int m   = tid >> 1;
            int sel = tid & 1;
            const float* src = A + (size_t)(m0 + m) * lda + kb + (sel << 4);
            if (kb + 32 < K) __builtin_prefetch(src + 32, 0, 3);
            __bf16 c0[8], c1[8];
            #pragma unroll
            for (int j = 0; j < 8; ++j) { c0[j] = (__bf16)src[j]; c1[j] = (__bf16)src[8 + j]; }
            // contiguous placement: [K0-7 ->0, K8-15 ->8, K16-23 ->16, K24-31 ->24]
            int p0 = sel ? 16 : 0;
            int p1 = sel ? 24 : 8;
            #pragma unroll
            for (int j = 0; j < 8; ++j) { As[m][p0 + j] = c0[j]; As[m][p1 + j] = c1[j]; }
        }
        // ---- stage B transposed into Bs[n][k]
        if (!btrans) {
            int n  = tid & 127;
            int kq = (tid >> 7) << 4;   // 0 or 16
            __bf16 t16[16];
            #pragma unroll
            for (int j = 0; j < 16; ++j)
                t16[j] = (__bf16)Bsrc[(size_t)(kb + kq + j) * ldb + n0 + n];
            #pragma unroll
            for (int j = 0; j < 16; ++j) Bs[n][kq + j] = t16[j];
        } else {
            int n  = tid >> 1;
            int kq = (tid & 1) << 4;
            const float* src = Bsrc + (size_t)(n0 + n) * ldb + kb + kq;
            if (kb + 32 < K) __builtin_prefetch(src + 32, 0, 3);
            #pragma unroll
            for (int j = 0; j < 16; ++j) Bs[n][kq + j] = (__bf16)src[j];
        }
        __syncthreads();

        // ---- fragment gathers: contiguous 32 B -> 2x ds_load_b128 each
        bfrag_u afrag[2], bfragv[4];
        #pragma unroll
        for (int mt = 0; mt < 2; ++mt) {
            const __bf16* p = &As[wm + (mt << 4) + lrow][lhalf];
            afrag[mt].h[0] = *(const bf16x8*)p;
            afrag[mt].h[1] = *(const bf16x8*)(p + 8);
        }
        #pragma unroll
        for (int nt = 0; nt < 4; ++nt) {
            const __bf16* p = &Bs[wn + (nt << 4) + lrow][lhalf];
            bfragv[nt].h[0] = *(const bf16x8*)p;
            bfragv[nt].h[1] = *(const bf16x8*)(p + 8);
        }

        #pragma unroll
        for (int mt = 0; mt < 2; ++mt)
            #pragma unroll
            for (int nt = 0; nt < 4; ++nt)
                acc[mt][nt] = __builtin_amdgcn_wmma_f32_16x16x32_bf16(
                    false, afrag[mt].v, false, bfragv[nt].v, (short)0,
                    acc[mt][nt], false, false);
        __syncthreads();
    }

    // ---- epilogue: C/D layout -> VGPR r: M = r + 8*(lane/16), N = lane%16
    #pragma unroll
    for (int mt = 0; mt < 2; ++mt) {
        int rbase = m0 + wm + (mt << 4) + ((lane >> 4) << 3);
        #pragma unroll
        for (int nt = 0; nt < 4; ++nt) {
            int col = n0 + wn + (nt << 4) + lrow;
            #pragma unroll
            for (int r = 0; r < 8; ++r)
                C[(size_t)(rbase + r) * ldc + col] = acc[mt][nt][r];
        }
    }
}

// =====================================================================
// Flash-style attention: one wave32 per (b, head, q-row).
// Online softmax; each lane owns hd/32 dims (d = i*32 + lane).
// win>0 -> sliding window; win==0 -> full causal. No score scaling
// (reference has none). KV heads repeated nh/kvh times.
// =====================================================================
__global__ __launch_bounds__(256) void attn_kernel(
    const float* __restrict__ q, const float* __restrict__ k,
    const float* __restrict__ v, float* __restrict__ out,
    const int* __restrict__ true_len, int nh, int kvh, int hd, int win)
{
    int wid  = blockIdx.x * 8 + (threadIdx.x >> 5);
    int lane = threadIdx.x & 31;
    int qpos = wid % Sm;
    int t    = wid / Sm;
    int hh   = t % nh;
    int b    = t / nh;
    int dpl  = hd >> 5;                 // 8 (HD=256) or 16 (GHD=512)
    int kvhead = hh / (nh / kvh);

    const float* qp = q + ((size_t)(b * Sm + qpos) * nh + hh) * hd;
    float qv[16], acc[16];
    for (int i = 0; i < dpl; ++i) { qv[i] = qp[i * 32 + lane]; acc[i] = 0.f; }

    float mcur = -3.0e38f, lsum = 0.f;
    int tl   = true_len[b];
    int kmax = qpos < tl - 1 ? qpos : tl - 1;
    int k0   = 0;
    if (win > 0) { k0 = qpos - win + 1; if (k0 < 0) k0 = 0; }

    for (int kk = k0; kk <= kmax; ++kk) {
        const float* kp = k + ((size_t)(b * Sm + kk) * kvh + kvhead) * hd;
        float p = 0.f;
        for (int i = 0; i < dpl; ++i) p += qv[i] * kp[i * 32 + lane];
        for (int off = 16; off > 0; off >>= 1) p += __shfl_xor(p, off, 32);

        float mnew = p > mcur ? p : mcur;
        float r = __expf(mcur - mnew);
        float e = __expf(p - mnew);
        lsum = lsum * r + e;
        const float* vp = v + ((size_t)(b * Sm + kk) * kvh + kvhead) * hd;
        for (int i = 0; i < dpl; ++i) acc[i] = acc[i] * r + e * vp[i * 32 + lane];
        mcur = mnew;
    }

    float inv = (lsum > 0.f) ? 1.0f / lsum : 0.f;
    float* op = out + ((size_t)(b * Sm + qpos) * nh + hh) * hd;
    for (int i = 0; i < dpl; ++i) op[i * 32 + lane] = acc[i] * inv;
}

// =====================================================================
// RMS norm: one block per row, LDS tree reduction. w==nullptr -> rms_ns.
// Safe in-place (each element read+written by the same thread).
// =====================================================================
__global__ __launch_bounds__(256) void rms_kernel(
    const float* __restrict__ x, const float* __restrict__ w,
    float* __restrict__ y, int d)
{
    int row = blockIdx.x;
    const float* xp = x + (size_t)row * d;
    float* yp       = y + (size_t)row * d;
    float s = 0.f;
    for (int i = threadIdx.x; i < d; i += 256) { float v = xp[i]; s += v * v; }
    __shared__ float red[256];
    red[threadIdx.x] = s;
    __syncthreads();
    for (int off = 128; off > 0; off >>= 1) {
        if (threadIdx.x < off) red[threadIdx.x] += red[threadIdx.x + off];
        __syncthreads();
    }
    float scale = rsqrtf(red[0] / (float)d + EPSf);
    for (int i = threadIdx.x; i < d; i += 256)
        yp[i] = xp[i] * scale * (w ? w[i] : 1.0f);
}

// Partial interleaved RoPE on x[rows, nh, hd], first `rot` dims.
__global__ void rope_kernel(float* __restrict__ x, int nh, int hd, int rot,
                            float theta, int total)
{
    int idx = blockIdx.x * blockDim.x + threadIdx.x;
    if (idx >= total) return;
    int hp  = rot >> 1;
    int j   = idx % hp;
    int t   = idx / hp;
    int hh  = t % nh;
    int row = t / nh;
    int s   = row % Sm;
    float inv = __powf(theta, -2.0f * (float)j / (float)rot);
    float ang = (float)s * inv;
    float c = __cosf(ang), sn = __sinf(ang);
    float* p = x + ((size_t)row * nh + hh) * hd + 2 * j;
    float x1 = p[0], x2 = p[1];
    p[0] = x1 * c - x2 * sn;
    p[1] = x1 * sn + x2 * c;
}

__global__ void embed_kernel(const int* __restrict__ tok,
                             const float* __restrict__ emb,
                             float* __restrict__ h)
{
    int row = blockIdx.x;
    const float* ep = emb + (size_t)tok[row] * Dm;
    float* hp = h + (size_t)row * Dm;
    const float sc = 45.254833995939045f;   // sqrt(2048)
    for (int i = threadIdx.x; i < Dm; i += 256) hp[i] = ep[i] * sc;
}

__global__ void add_kernel(float* __restrict__ a, const float* __restrict__ b, int n)
{
    int i = blockIdx.x * blockDim.x + threadIdx.x;
    if (i < n) a[i] += b[i];
}

__global__ void add_scale_kernel(float* __restrict__ a, const float* __restrict__ b,
                                 const float* __restrict__ sc, int n)
{
    int i = blockIdx.x * blockDim.x + threadIdx.x;
    if (i < n) a[i] = (a[i] + b[i]) * sc[0];
}

__global__ void gelu_mul_kernel(float* __restrict__ g, const float* __restrict__ u, int n)
{
    int i = blockIdx.x * blockDim.x + threadIdx.x;
    if (i < n) {
        float x = g[i];
        float t = 0.5f * x * (1.0f + tanhf(0.7978845608028654f * (x + 0.044715f * x * x * x)));
        g[i] = t * u[i];
    }
}

// =====================================================================
// Host orchestration
// =====================================================================
extern "C" void kernel_launch(void* const* d_in, const int* in_sizes, int n_in,
                              void* d_out, int out_size, void* d_ws, size_t ws_size,
                              hipStream_t stream)
{
    const int*   tokens = (const int*)  d_in[0];
    const int*   tlen   = (const int*)  d_in[1];
    const float* embed  = (const float*)d_in[2];
    const float* norm_w = (const float*)d_in[3];
    const float* s_wq = (const float*)d_in[4];
    const float* s_wk = (const float*)d_in[5];
    const float* s_wv = (const float*)d_in[6];
    const float* s_wo = (const float*)d_in[7];
    const float* s_qn = (const float*)d_in[8];
    const float* s_kn = (const float*)d_in[9];
    const float* s_wg = (const float*)d_in[10];
    const float* s_wu = (const float*)d_in[11];
    const float* s_wd = (const float*)d_in[12];
    const float* s_ln1 = (const float*)d_in[13];
    const float* s_ln2 = (const float*)d_in[14];
    const float* s_ln3 = (const float*)d_in[15];
    const float* s_ln4 = (const float*)d_in[16];
    const float* s_scal = (const float*)d_in[17];
    const float* g_wq = (const float*)d_in[18];
    const float* g_wk = (const float*)d_in[19];
    const float* g_wo = (const float*)d_in[20];
    const float* g_qn = (const float*)d_in[21];
    const float* g_kn = (const float*)d_in[22];
    const float* g_wg = (const float*)d_in[23];
    const float* g_wu = (const float*)d_in[24];
    const float* g_wd = (const float*)d_in[25];
    const float* g_ln1 = (const float*)d_in[26];
    const float* g_ln2 = (const float*)d_in[27];
    const float* g_ln3 = (const float*)d_in[28];
    const float* g_ln4 = (const float*)d_in[29];
    const float* g_scal = (const float*)d_in[30];

    // ---- workspace carve (floats) ----
    const size_t MD = (size_t)MROWS * Dm;        // 4M
    float* ws   = (float*)d_ws;
    float* hbuf = ws;                 // h residual stream          4M
    float* hn   = hbuf + MD;          // normed activations         4M
    float* qb   = hn   + MD;          // q  (up to 2048x4096)       8M
    float* kb   = qb   + 2 * MD;      // k  (up to 2048x2048)       4M
    float* vb   = kb   + MD;          // v                          4M
    float* attb = vb   + MD;          // attn out (up to 2048x4096) 8M
    float* tmp  = attb + 2 * MD;      // proj / ffn-down out        4M
    float* gb   = tmp  + MD;          // gate  2048x8192           16M
    float* ub   = gb   + (size_t)MROWS * Fm;  // up                16M

    auto gemm = [&](const float* A, const float* B, float* C,
                    int M, int N, int K, int btrans, int ldb) {
        dim3 grid(N / 128, M / 128);
        gemm_bf16_wmma_kernel<<<grid, 256, 0, stream>>>(A, B, C, M, N, K, K, ldb, N, btrans);
    };

    const int M = MROWS;
    embed_kernel<<<M, 256, 0, stream>>>(tokens, embed, hbuf);

    auto run_block = [&](const float* wq, const float* wk, const float* wv,
                         const float* wo, const float* qn, const float* kn,
                         const float* wg, const float* wu, const float* wd,
                         const float* ln1, const float* ln2, const float* ln3,
                         const float* ln4, const float* scal,
                         int hd, int kvh, int rot, float theta, int win) {
        const int NQ = Hn * hd, NK = kvh * hd;
        rms_kernel<<<M, 256, 0, stream>>>(hbuf, ln1, hn, Dm);
        gemm(hn, wq, qb, M, NQ, Dm, 0, NQ);
        gemm(hn, wk, kb, M, NK, Dm, 0, NK);
        if (wv) {
            gemm(hn, wv, vb, M, NK, Dm, 0, NK);
            rms_kernel<<<M * kvh, 256, 0, stream>>>(vb, nullptr, vb, hd);
        } else {
            // v = rms_ns(raw k) before kn-norm / rope
            rms_kernel<<<M * kvh, 256, 0, stream>>>(kb, nullptr, vb, hd);
        }
        rms_kernel<<<M * Hn,  256, 0, stream>>>(qb, qn, qb, hd);
        rms_kernel<<<M * kvh, 256, 0, stream>>>(kb, kn, kb, hd);
        int tq = M * Hn  * (rot / 2);
        int tk = M * kvh * (rot / 2);
        rope_kernel<<<(tq + 255) / 256, 256, 0, stream>>>(qb, Hn,  hd, rot, theta, tq);
        rope_kernel<<<(tk + 255) / 256, 256, 0, stream>>>(kb, kvh, hd, rot, theta, tk);
        attn_kernel<<<(Bm_ * Hn * Sm) / 8, 256, 0, stream>>>(qb, kb, vb, attb, tlen,
                                                             Hn, kvh, hd, win);
        gemm(attb, wo, tmp, M, Dm, NQ, 0, Dm);
        rms_kernel<<<M, 256, 0, stream>>>(tmp, ln2, tmp, Dm);
        add_kernel<<<(M * Dm) / 256, 256, 0, stream>>>(hbuf, tmp, M * Dm);
        // FFN
        rms_kernel<<<M, 256, 0, stream>>>(hbuf, ln3, hn, Dm);
        gemm(hn, wg, gb, M, Fm, Dm, 0, Fm);
        gemm(hn, wu, ub, M, Fm, Dm, 0, Fm);
        gelu_mul_kernel<<<(M * Fm) / 256, 256, 0, stream>>>(gb, ub, M * Fm);
        gemm(gb, wd, tmp, M, Dm, Fm, 0, Dm);
        rms_kernel<<<M, 256, 0, stream>>>(tmp, ln4, tmp, Dm);
        add_scale_kernel<<<(M * Dm) / 256, 256, 0, stream>>>(hbuf, tmp, scal, M * Dm);
    };

    // Block 1: sliding window, full RoPE (rot=HD), theta=1e4
    run_block(s_wq, s_wk, s_wv, s_wo, s_qn, s_kn, s_wg, s_wu, s_wd,
              s_ln1, s_ln2, s_ln3, s_ln4, s_scal,
              HDs, KVs, HDs, 10000.0f, WINs);
    // Block 2: global causal, partial RoPE (rot=128), theta=1e6, v=rms_ns(k)
    run_block(g_wq, g_wk, nullptr, g_wo, g_qn, g_kn, g_wg, g_wu, g_wd,
              g_ln1, g_ln2, g_ln3, g_ln4, g_scal,
              GHD, GKV, ROTg, 1000000.0f, 0);

    // Final norm + tied-embedding logits (B transposed: embed is V x D)
    rms_kernel<<<M, 256, 0, stream>>>(hbuf, norm_w, hn, Dm);
    gemm(hn, embed, (float*)d_out, M, Vm, Dm, /*btrans=*/1, /*ldb=*/Dm);
}
